// SpatioTemporalSSM_46574625358438
// MI455X (gfx1250) — matrix operand order repaired and checked by reading, not compiled, not run on hardware
//
#include <hip/hip_runtime.h>
#include <hip/hip_bf16.h>

// ---------------------------------------------------------------------------
// SpatioTemporal SSM for MI455X (gfx1250), bf16 WMMA with fp32 accumulate.
// Gating uses branch-free hardware transcendentals (v_exp_f32 / v_rcp_f32 /
// v_tanh_f32) so no EXEC-modifying slow paths appear near the WMMA region.
// ---------------------------------------------------------------------------

typedef __attribute__((ext_vector_type(16))) __bf16 v16bf;
typedef __attribute__((ext_vector_type(8)))  float  v8f;

union Frag {
    v16bf v;
    float4 f[2];
};

// Geometry constants of the reference problem.
#define BB 8
#define TT 16
#define DD 128
#define HH 32
#define WW 32
#define TOK_PER_FRAME (DD * HH * WW)         // 131072 floats per (b,t) frame
#define TOK_TOTAL ((long)BB * TT * HH * WW)  // 131072 spatial-temporal tokens
#define ELEMS ((long)BB * TT * DD * HH * WW) // 16777216

#define NT 128          // tokens per block (4 seqs x 32 or 8 seqs x 16)
#define NMT 8           // M-tiles per block (NT/16)

#define LOG2E 1.4426950408889634f

// ---- branch-free transcendentals (single VALU trans ops, full range) ------
__device__ __forceinline__ float exp2_fast(float x) {
    return __builtin_amdgcn_exp2f(x);        // v_exp_f32
}
__device__ __forceinline__ float rcp_fast(float x) {
    return __builtin_amdgcn_rcpf(x);         // v_rcp_f32
}
__device__ __forceinline__ float sigmoid_fast(float x) {
    // exp2(-inf)=0 -> 1 ; exp2(+inf)=inf -> rcp(inf)=0. No branches.
    return rcp_fast(1.0f + exp2_fast(-LOG2E * x));
}
__device__ __forceinline__ float tanh_fast(float x) {
#if __has_builtin(__builtin_amdgcn_tanhf)
    return __builtin_amdgcn_tanhf(x);        // native v_tanh_f32 on gfx1250
#else
    // tanh(x) = 1 - 2/(exp2(2*log2e*x)+1); exact +/-1 limits, branch-free.
    return 1.0f - 2.0f * rcp_fast(exp2_fast(2.0f * LOG2E * x) + 1.0f);
#endif
}

// token -> global float offset (excluding channel d * 1024)
__device__ __forceinline__ long tok_off(int mode, int seqG, int l) {
    if (mode <= 1) {                 // horizontal rows (lr / rl)
        int bt = seqG >> 5, y = seqG & 31;
        int xc = (mode == 1) ? (31 - l) : l;
        return (long)bt * TOK_PER_FRAME + y * WW + xc;
    } else if (mode <= 3) {          // vertical cols (ud / du)
        int bt = seqG >> 5, xc = seqG & 31;
        int yy = (mode == 3) ? (31 - l) : l;
        return (long)bt * TOK_PER_FRAME + yy * WW + xc;
    } else {                          // temporal: seq = (b, y*32+x), l = t
        int b = seqG >> 10, p = seqG & 1023;
        return (long)b * (TT * (long)TOK_PER_FRAME) + (long)l * TOK_PER_FRAME + p;
    }
}

// Fused SSM pass: z = x*Wi^T ; c = sig(i)*tanh(u) ; og = sig(o) ;
// scan s = a*s + c ; y = og*s ; out = y*Wo^T ; dst (=|+=) out.
__global__ void __launch_bounds__(256)
ssm_pass(const float* __restrict__ src, float* __restrict__ dst,
         const unsigned short* __restrict__ WiU,
         const unsigned short* __restrict__ WoU,
         const float* __restrict__ adecay,
         int mode, int lshift, int seqPerBlk, float inScale, int firstWrite) {
    extern __shared__ char smem[];
    __bf16* sh_x = (__bf16*)smem;                       // [NT][128] inputs
    __bf16* sh_c = (__bf16*)(smem + NT * DD * 2);       // [NT][128] c -> y
    __bf16* sh_o = (__bf16*)(smem + 2 * NT * DD * 2);   // [NT][128] out gate
    const __bf16* Wi = reinterpret_cast<const __bf16*>(WiU);   // [384][128]
    const __bf16* Wo = reinterpret_cast<const __bf16*>(WoU);   // [128][128]

    const int tid  = threadIdx.x;
    const int wave = tid >> 5;
    const int lane = tid & 31;
    const int L    = 1 << lshift;
    const int seq0 = blockIdx.x * seqPerBlk;

    // ---------------- Phase 0: gather tokens into LDS (bf16) --------------
    for (int idx = tid; idx < NT * DD; idx += 256) {
        int t = idx >> 7, d = idx & 127;
        int s = t >> lshift, l = t & (L - 1);
        long g = tok_off(mode, seq0 + s, l) + (long)d * (HH * WW);
        sh_x[idx] = (__bf16)(src[g] * inScale);
    }
    __syncthreads();

    // Fragment lane decomposition (ISA 7.12.2 layouts).
    const int n  = lane & 15;               // A: row M ; B/C: column N
    const int kh = (lane & 16) ? 8 : 0;     // A-frag K sub-offset
    const int kb = (lane & 16) ? 16 : 0;    // unused; B-frag K offset is kh, as for A
    const int mo = (lane & 16) ? 8 : 0;     // C-frag M offset
    const int dbase = wave * 16;            // this wave's channel group

    // ---------------- Phase A: input projection + gating ------------------
    for (int mt = 0; mt < NMT; ++mt) {
        const int tb = mt * 16;
        Frag afr[4];
        const __bf16* xrow = sh_x + (tb + n) * DD;
#pragma unroll
        for (int ks = 0; ks < 4; ++ks) {
            int k0 = ks * 32 + kh;
            afr[ks].f[0] = *(const float4*)(xrow + k0);
            afr[ks].f[1] = *(const float4*)(xrow + k0 + 16);
        }
        v8f accU = {}, accI = {}, accO = {};
#pragma unroll
        for (int g = 0; g < 3; ++g) {
            const __bf16* wrow = Wi + (g * DD + dbase + n) * DD;
            v8f acc = {};
#pragma unroll
            for (int ks = 0; ks < 4; ++ks) {
                Frag bfr;
                const __bf16* wp = wrow + ks * 32 + kh;
                bfr.f[0] = *(const float4*)(wp);
                bfr.f[1] = *(const float4*)(wp + 16);
                acc = __builtin_amdgcn_wmma_f32_16x16x32_bf16(
                    false, afr[ks].v, false, bfr.v, (short)0, acc, false, false);
            }
            if (g == 0) accU = acc; else if (g == 1) accI = acc; else accO = acc;
        }
#pragma unroll
        for (int j = 0; j < 8; ++j) {
            float c  = sigmoid_fast(accI[j]) * tanh_fast(accU[j]);
            float og = sigmoid_fast(accO[j]);
            int row = tb + mo + j;
            sh_c[row * DD + dbase + n] = (__bf16)c;
            sh_o[row * DD + dbase + n] = (__bf16)og;
        }
    }
    __syncthreads();

    // ---------------- Phase B: channel-wise linear recurrence -------------
    const int units = seqPerBlk * DD;
    for (int u = tid; u < units; u += 256) {
        int s = u >> 7, d = u & 127;
        float a  = adecay[d];
        float st = 0.0f;
        int base = (s << lshift) * DD + d;
        for (int l = 0; l < L; ++l) {
            int off = base + l * DD;
            st = a * st + (float)sh_c[off];
            sh_c[off] = (__bf16)(st * (float)sh_o[off]);
        }
    }
    __syncthreads();

    // ---------------- Phase C: output projection + global accumulate ------
    const int ebase = wave * 16;
    for (int mt = 0; mt < NMT; ++mt) {
        const int tb = mt * 16;
        const __bf16* yrow = sh_c + (tb + n) * DD;
        const __bf16* wrow = Wo + (ebase + n) * DD;
        v8f acc = {};
#pragma unroll
        for (int ks = 0; ks < 4; ++ks) {
            Frag afr, bfr;
            afr.f[0] = *(const float4*)(yrow + ks * 32 + kh);
            afr.f[1] = *(const float4*)(yrow + ks * 32 + kh + 16);
            bfr.f[0] = *(const float4*)(wrow + ks * 32 + kh);
            bfr.f[1] = *(const float4*)(wrow + ks * 32 + kh + 16);
            acc = __builtin_amdgcn_wmma_f32_16x16x32_bf16(
                false, afr.v, false, bfr.v, (short)0, acc, false, false);
        }
#pragma unroll
        for (int j = 0; j < 8; ++j) {
            int row = tb + mo + j;
            int s = row >> lshift, l = row & (L - 1);
            long g = tok_off(mode, seq0 + s, l) + (long)(ebase + n) * (HH * WW);
            if (firstWrite) dst[g] = acc[j];
            else            dst[g] += acc[j];
        }
    }
}

// ---------------------------------------------------------------------------
__global__ void cvt_bf16_kernel(const float* __restrict__ src,
                                unsigned short* __restrict__ dst, int nelem) {
    int i = blockIdx.x * blockDim.x + threadIdx.x;
    if (i < nelem) {
        __bf16 h = (__bf16)src[i];
        dst[i] = *reinterpret_cast<unsigned short*>(&h);
    }
}

__global__ void decay_kernel(const float* __restrict__ A,
                             float* __restrict__ a, int nelem) {
    int i = blockIdx.x * blockDim.x + threadIdx.x;
    if (i < nelem) {
        float e = exp2_fast(LOG2E * A[i]);   // exp(A), branch-free
        a[i] = e < 0.999f ? e : 0.999f;
    }
}

// One wave32 per token; LayerNorm over D=128 (stride h*w), in place.
__global__ void __launch_bounds__(256)
ln_kernel(float* __restrict__ x, const float* __restrict__ g,
          const float* __restrict__ b) {
    int wv   = (blockIdx.x * blockDim.x + threadIdx.x) >> 5;
    int lane = threadIdx.x & 31;
    if (wv >= (int)TOK_TOTAL) return;
    int bt = wv >> 10, p = wv & 1023;
    long base = (long)bt * TOK_PER_FRAME + p;
    float v[4], sum = 0.0f, sq = 0.0f;
#pragma unroll
    for (int j = 0; j < 4; ++j) {
        int d = lane * 4 + j;
        v[j] = x[base + (long)d * (HH * WW)];
        sum += v[j];
        sq  += v[j] * v[j];
    }
#pragma unroll
    for (int off = 16; off > 0; off >>= 1) {
        sum += __shfl_xor(sum, off, 32);
        sq  += __shfl_xor(sq,  off, 32);
    }
    float mu  = sum * (1.0f / 128.0f);
    float var = sq * (1.0f / 128.0f) - mu * mu;
    float inv = rsqrtf(var + 1e-5f);
#pragma unroll
    for (int j = 0; j < 4; ++j) {
        int d = lane * 4 + j;
        x[base + (long)d * (HH * WW)] = (v[j] - mu) * inv * g[d] + b[d];
    }
}

// ---------------------------------------------------------------------------
extern "C" void kernel_launch(void* const* d_in, const int* in_sizes, int n_in,
                              void* d_out, int out_size, void* d_ws, size_t ws_size,
                              hipStream_t stream) {
    (void)in_sizes; (void)n_in; (void)out_size; (void)ws_size;
    const float* tok   = (const float*)d_in[0];
    const float* s_inp = (const float*)d_in[1];
    const float* s_A   = (const float*)d_in[2];
    const float* s_out = (const float*)d_in[3];
    const float* t_inp = (const float*)d_in[4];
    const float* t_A   = (const float*)d_in[5];
    const float* t_out = (const float*)d_in[6];
    const float* ln_g  = (const float*)d_in[7];
    const float* ln_b  = (const float*)d_in[8];

    float* xbuf = (float*)d_out;                 // live activation buffer
    char*  ws   = (char*)d_ws;
    float* bufY = (float*)ws;                    // 4-direction spatial sum
    size_t off = (size_t)ELEMS * sizeof(float);
    unsigned short* WiS = (unsigned short*)(ws + off); off += 4ull * 384 * 128 * 2;
    unsigned short* WoS = (unsigned short*)(ws + off); off += 4ull * 128 * 128 * 2;
    unsigned short* WiT = (unsigned short*)(ws + off); off += 4ull * 384 * 128 * 2;
    unsigned short* WoT = (unsigned short*)(ws + off); off += 4ull * 128 * 128 * 2;
    float* aS = (float*)(ws + off); off += 4ull * 128 * sizeof(float);
    float* aT = (float*)(ws + off);

    // x <- tok
    hipMemcpyAsync(xbuf, tok, (size_t)ELEMS * sizeof(float),
                   hipMemcpyDeviceToDevice, stream);

    // Weight conversion fp32 -> bf16, decay precompute.
    cvt_bf16_kernel<<<(196608 + 255) / 256, 256, 0, stream>>>(s_inp, WiS, 196608);
    cvt_bf16_kernel<<<(65536  + 255) / 256, 256, 0, stream>>>(s_out, WoS, 65536);
    cvt_bf16_kernel<<<(196608 + 255) / 256, 256, 0, stream>>>(t_inp, WiT, 196608);
    cvt_bf16_kernel<<<(65536  + 255) / 256, 256, 0, stream>>>(t_out, WoT, 65536);
    decay_kernel<<<2, 256, 0, stream>>>(s_A, aS, 512);
    decay_kernel<<<2, 256, 0, stream>>>(t_A, aT, 512);

    const size_t lds = 3ull * NT * DD * 2;       // 96 KB dynamic LDS

    for (int l = 0; l < 4; ++l) {
        const unsigned short* Wi = WiS + (size_t)l * 384 * 128;
        const unsigned short* Wo = WoS + (size_t)l * 128 * 128;
        const float* ad = aS + l * 128;
        // 4 spatial directional scans: bufY (=,+=) ssm_dir(x); avg folded later.
        for (int dir = 0; dir < 4; ++dir) {
            ssm_pass<<<1024, 256, lds, stream>>>(
                xbuf, bufY, Wi, Wo, ad,
                /*mode=*/dir, /*lshift=*/5, /*seqPerBlk=*/4,
                /*inScale=*/1.0f, /*firstWrite=*/dir == 0);
        }
        // Temporal scan on xs = bufY/4, residual: x += ssm_t(xs).
        ssm_pass<<<1024, 256, lds, stream>>>(
            bufY, xbuf,
            WiT + (size_t)l * 384 * 128, WoT + (size_t)l * 128 * 128,
            aT + l * 128,
            /*mode=*/4, /*lshift=*/4, /*seqPerBlk=*/8,
            /*inScale=*/0.25f, /*firstWrite=*/0);
    }

    // Final LayerNorm over D, in place on d_out.
    ln_kernel<<<16384, 256, 0, stream>>>(xbuf, ln_g, ln_b);
}
